// MultiHeadAttention_11227044512296
// MI455X (gfx1250) — matrix operand order repaired and checked by reading, not compiled, not run on hardware
//
#include <hip/hip_runtime.h>

// ---------------------------------------------------------------------------
// Multi-head attention forward for MI455X (gfx1250, wave32, WMMA).
//   convert f32 -> bf16 (weights once; q/k/v via reused staging buffer)
//   3x GEMM  (bf16 WMMA, f32 accum, bias, bf16 out)
//   flash attention (bf16 WMMA, online softmax via v_permlane16 butterflies,
//                    denominator via P@ones WMMA, double-buffered V^T in LDS)
//   1x GEMM  (bf16 WMMA, bias, f32 out)
// ---------------------------------------------------------------------------

#define EMB    1024
#define NHEAD  16
#define HDIM   64
#define BATCH  2
#define SEQ    2048
#define MROWS  (BATCH * SEQ)          // 4096 rows for all projection GEMMs
#define LOG2E  1.44269504088896340736f
#define SCALE  0.03125f               // 1/sqrt(EMB) = 1/32

typedef __bf16 bf16_t;
typedef __attribute__((ext_vector_type(8)))  __bf16 v8bf;
typedef __attribute__((ext_vector_type(16))) __bf16 v16bf;
typedef __attribute__((ext_vector_type(4)))  float  v4f;
typedef __attribute__((ext_vector_type(8)))  float  v8f;

static __device__ inline v8f zero_v8f() {
    v8f z;
#pragma unroll
    for (int i = 0; i < 8; ++i) z[i] = 0.0f;
    return z;
}

static __device__ inline v16bf cat16(v8bf lo, v8bf hi) {
    return __builtin_shufflevector(lo, hi, 0, 1, 2, 3, 4, 5, 6, 7,
                                           8, 9, 10, 11, 12, 13, 14, 15);
}

// ---------------------------------------------------------------------------
// Row-max butterfly within each 16-lane group.
// v_permlane16_b32 (VALU, co-executes with the matrix pipe; no LDS waits)
// with nibble selectors encoding lane^1 / lane^2 / lane^4 / lane^8.
// ---------------------------------------------------------------------------
#if __has_builtin(__builtin_amdgcn_permlane16)
static __device__ inline float xor16_max(float v) {
    unsigned x, y;
    x = __builtin_bit_cast(unsigned, v);
    y = __builtin_amdgcn_permlane16(x, x, 0x67452301u, 0xEFCDAB89u, false, false);
    v = fmaxf(v, __builtin_bit_cast(float, y));
    x = __builtin_bit_cast(unsigned, v);
    y = __builtin_amdgcn_permlane16(x, x, 0x54761032u, 0xDCFE98BAu, false, false);
    v = fmaxf(v, __builtin_bit_cast(float, y));
    x = __builtin_bit_cast(unsigned, v);
    y = __builtin_amdgcn_permlane16(x, x, 0x32107654u, 0xBA98FEDCu, false, false);
    v = fmaxf(v, __builtin_bit_cast(float, y));
    x = __builtin_bit_cast(unsigned, v);
    y = __builtin_amdgcn_permlane16(x, x, 0xFEDCBA98u, 0x76543210u, false, false);
    v = fmaxf(v, __builtin_bit_cast(float, y));
    return v;
}
#else
static __device__ inline float xor16_max(float v) {
    v = fmaxf(v, __shfl_xor(v, 1, 32));
    v = fmaxf(v, __shfl_xor(v, 2, 32));
    v = fmaxf(v, __shfl_xor(v, 4, 32));
    v = fmaxf(v, __shfl_xor(v, 8, 32));
    return v;
}
#endif

// ---------------------------------------------------------------------------
// f32 -> bf16 conversion pass (element counts are multiples of 8)
// ---------------------------------------------------------------------------
__global__ __launch_bounds__(256) void cvt_bf16_kernel(
    const float* __restrict__ src, bf16_t* __restrict__ dst, int n) {
    const int i = (blockIdx.x * 256 + threadIdx.x) * 8;
    if (i >= n) return;
    v4f a = *(const v4f*)(src + i);
    v4f b = *(const v4f*)(src + i + 4);
    v8bf r;
#pragma unroll
    for (int j = 0; j < 4; ++j) r[j]     = (__bf16)a[j];
#pragma unroll
    for (int j = 0; j < 4; ++j) r[4 + j] = (__bf16)b[j];
    *(v8bf*)(dst + i) = r;
}

// ---------------------------------------------------------------------------
// WMMA fragment loaders (layouts per CDNA5 ISA 7.12.2, 16-bit, wave32)
//   A 16x32 (MxK): lanes 0-15 row M=lane, K {k0..k0+7, k0+16..k0+23};
//                  lanes 16-31 row M=lane-16, K {k0+8..15, k0+24..31}.
//   B 32x16 (KxN): lanes 0-15 col N=lane,  K {k0..k0+7, k0+16..k0+23};
//                  lanes 16-31 col N=lane-16, K {k0+8..15, k0+24..31}.
// ---------------------------------------------------------------------------
static __device__ inline v16bf load_a_bf16(const bf16_t* __restrict__ X, int ld,
                                           int m0, int k0, int lane) {
    const int half = lane >> 4;
    const bf16_t* p = X + (size_t)(m0 + (lane & 15)) * ld + k0 + half * 8;
    return cat16(*(const v8bf*)p, *(const v8bf*)(p + 16));
}

// B fragment = (rows of bf16 matrix)^T: lane n reads row n0+n contiguously.
static __device__ inline v16bf load_bt_bf16(const bf16_t* __restrict__ R, int ld,
                                            int n0, int k0, int lane) {
    const int half = lane >> 4;
    const bf16_t* p = R + (size_t)(n0 + (lane & 15)) * ld + k0 + half * 8;
    return cat16(*(const v8bf*)p, *(const v8bf*)(p + 16));
}

// ---------------------------------------------------------------------------
// GEMM: Y[M,N] = X[M,K](bf16) @ W[N,K]^T(bf16) + bias(f32), out bf16 or f32.
// 256 threads = 8 waves; block tile 128x128; wave tile 64Mx32N (4x2 WMMA).
// All dims are multiples of 128 (M=4096, N=K=1024): no bounds checks.
// ---------------------------------------------------------------------------
template <bool OUT_BF16>
__global__ __launch_bounds__(256) void gemm_xwt_kernel(
    const bf16_t* __restrict__ X, const bf16_t* __restrict__ W,
    const float* __restrict__ bias, bf16_t* __restrict__ Yb,
    float* __restrict__ Yf, int M, int N, int K) {
    const int lane = threadIdx.x & 31;
    const int wave = threadIdx.x >> 5;
    const int m0 = blockIdx.y * 128 + (wave >> 2) * 64;   // 4 M-tiles of 16
    const int n0 = blockIdx.x * 128 + (wave & 3) * 32;    // 2 N-tiles of 16

    v8f acc[4][2];
#pragma unroll
    for (int i = 0; i < 4; ++i)
#pragma unroll
        for (int j = 0; j < 2; ++j) acc[i][j] = zero_v8f();

#pragma unroll 2
    for (int k0 = 0; k0 < K; k0 += 32) {
        v16bf a[4], b[2];
#pragma unroll
        for (int i = 0; i < 4; ++i) a[i] = load_a_bf16(X, K, m0 + 16 * i, k0, lane);
#pragma unroll
        for (int j = 0; j < 2; ++j) b[j] = load_bt_bf16(W, K, n0 + 16 * j, k0, lane);
        if (k0 + 64 < K) {   // prefetch next-next k-step (global_prefetch_b8)
            __builtin_prefetch(X + (size_t)(m0 + (lane & 15)) * K + k0 + 64, 0, 0);
            __builtin_prefetch(W + (size_t)(n0 + (lane & 15)) * K + k0 + 64, 0, 0);
        }
#pragma unroll
        for (int i = 0; i < 4; ++i)
#pragma unroll
            for (int j = 0; j < 2; ++j)
                acc[i][j] = __builtin_amdgcn_wmma_f32_16x16x32_bf16(
                    false, a[i], false, b[j], (short)0, acc[i][j], false, false);
    }

    const float bc0 = bias[n0 + (lane & 15)];
    const float bc1 = bias[n0 + 16 + (lane & 15)];
    const int rhalf = (lane >> 4) * 8;
#pragma unroll
    for (int i = 0; i < 4; ++i) {
#pragma unroll
        for (int j = 0; j < 2; ++j) {
            const float bc = (j == 0) ? bc0 : bc1;
            const int col = n0 + 16 * j + (lane & 15);
#pragma unroll
            for (int r = 0; r < 8; ++r) {
                const int row = m0 + 16 * i + r + rhalf;
                const float v = acc[i][j][r] + bc;
                if (OUT_BF16) Yb[(size_t)row * N + col] = (bf16_t)v;
                else          Yf[(size_t)row * N + col] = v;
            }
        }
    }
}

// ---------------------------------------------------------------------------
// Flash attention: one block per (b, h, 64-row Q tile); 4 waves, each wave
// streams 16 Q rows against all 2048 keys in 32-key tiles.
// Double-buffered V^T tiles in LDS (one barrier per tile, staging overlaps
// the WMMA stream). Row max via permlane16 butterflies; row sums via a
// P@ones WMMA; P re-laid-out through a per-wave LDS bounce.
// ---------------------------------------------------------------------------
#define VT_STRIDE 40   // ushort per Vt row (32 keys + pad, 80B, 16B-aligned)
#define PL_STRIDE 40   // ushort per P row

// Transpose one 32-key x 64-dim V tile into dst[dim][key] (whole block).
static __device__ inline void stage_vt(unsigned short* __restrict__ dst,
                                       const bf16_t* __restrict__ Vh,
                                       int kk, int tid) {
    const int rp = tid & 15;     // key-pair 0..15
    const int cg = tid >> 4;     // dim group 0..7 (8 dims each)
    const bf16_t* p0 = Vh + (size_t)(kk + 2 * rp) * EMB + cg * 8;
    v8bf r0 = *(const v8bf*)p0;
    v8bf r1 = *(const v8bf*)(p0 + EMB);
#pragma unroll
    for (int c = 0; c < 8; ++c) {
        unsigned int u0 = __builtin_bit_cast(unsigned short, r0[c]);
        unsigned int u1 = __builtin_bit_cast(unsigned short, r1[c]);
        *(unsigned int*)&dst[(cg * 8 + c) * VT_STRIDE + 2 * rp] = u0 | (u1 << 16);
    }
}

__global__ __launch_bounds__(128) void attention_kernel(
    const bf16_t* __restrict__ Qp, const bf16_t* __restrict__ Kp,
    const bf16_t* __restrict__ Vp, bf16_t* __restrict__ Ctx) {
    const int lane = threadIdx.x & 31;
    const int wave = threadIdx.x >> 5;      // 0..3
    const int b = blockIdx.z;
    const int h = blockIdx.y;
    const int q0 = blockIdx.x * 64 + wave * 16;
    const size_t rowbase = (size_t)b * SEQ;

    const bf16_t* Qh = Qp + rowbase * EMB + h * HDIM;
    const bf16_t* Kh = Kp + rowbase * EMB + h * HDIM;
    const bf16_t* Vh = Vp + rowbase * EMB + h * HDIM;

    __shared__ unsigned short VtU[2][HDIM * VT_STRIDE];  // double-buffered V^T
    __shared__ unsigned short Pl[4][16 * PL_STRIDE];     // per-wave P scratch

    // Q strip lives in registers for the whole pass: two A fragments (K=0..63).
    const v16bf qa0 = load_a_bf16(Qh, EMB, q0, 0, lane);
    const v16bf qa1 = load_a_bf16(Qh, EMB, q0, 32, lane);

    // all-ones B fragment for the row-sum WMMA
    v16bf ones;
#pragma unroll
    for (int i = 0; i < 16; ++i) ones[i] = (__bf16)1.0f;

    v8f acc[4];
#pragma unroll
    for (int t = 0; t < 4; ++t) acc[t] = zero_v8f();
    v8f acc_l = zero_v8f();                 // softmax denominators (replicated)
    float m_run[8];
#pragma unroll
    for (int r = 0; r < 8; ++r) m_run[r] = -3.0e38f;

    stage_vt(VtU[0], Vh, 0, threadIdx.x);   // preload tile 0

    for (int it = 0; it < SEQ / 32; ++it) {
        const int kk = it * 32;
        const int p = it & 1;
        __syncthreads();   // tile `it` staged; previous compute (same buf) done
        if (kk + 32 < SEQ)
            stage_vt(VtU[1 - p], Vh, kk + 32, threadIdx.x);  // overlap staging
        const unsigned short* Vt = VtU[p];

        // ---- scores: s[j] = Q(16x64) @ K^T(64x16) for key tiles kk, kk+16 --
        v8f s0 = zero_v8f(), s1 = zero_v8f();
        {
            v16bf b00 = load_bt_bf16(Kh, EMB, kk,      0,  lane);
            v16bf b01 = load_bt_bf16(Kh, EMB, kk,      32, lane);
            v16bf b10 = load_bt_bf16(Kh, EMB, kk + 16, 0,  lane);
            v16bf b11 = load_bt_bf16(Kh, EMB, kk + 16, 32, lane);
            if (kk + 32 < SEQ)
                __builtin_prefetch(Kh + (size_t)(kk + 32 + lane) * EMB, 0, 0);
            s0 = __builtin_amdgcn_wmma_f32_16x16x32_bf16(false, qa0, false, b00,
                                                         (short)0, s0, false, false);
            s0 = __builtin_amdgcn_wmma_f32_16x16x32_bf16(false, qa1, false, b01,
                                                         (short)0, s0, false, false);
            s1 = __builtin_amdgcn_wmma_f32_16x16x32_bf16(false, qa0, false, b10,
                                                         (short)0, s1, false, false);
            s1 = __builtin_amdgcn_wmma_f32_16x16x32_bf16(false, qa1, false, b11,
                                                         (short)0, s1, false, false);
        }

        // ---- online softmax: row max butterfly (permlane16, VALU-only) -----
        float corr[8];
#pragma unroll
        for (int r = 0; r < 8; ++r) {
            s0[r] *= SCALE;
            s1[r] *= SCALE;
            const float v = xor16_max(fmaxf(s0[r], s1[r]));
            const float mnew = fmaxf(m_run[r], v);
            corr[r] = __builtin_amdgcn_exp2f((m_run[r] - mnew) * LOG2E);
            m_run[r] = mnew;
            s0[r] = __builtin_amdgcn_exp2f((s0[r] - mnew) * LOG2E);
            s1[r] = __builtin_amdgcn_exp2f((s1[r] - mnew) * LOG2E);
        }
#pragma unroll
        for (int t = 0; t < 4; ++t)
#pragma unroll
            for (int r = 0; r < 8; ++r) acc[t][r] *= corr[r];
#pragma unroll
        for (int r = 0; r < 8; ++r) acc_l[r] *= corr[r];

        // ---- C-layout P -> LDS -> A-layout fragment (per-wave scratch) ----
        {
            unsigned short* Pw = Pl[wave];
            const int chalf = (lane >> 4) * 8;
#pragma unroll
            for (int r = 0; r < 8; ++r) {
                const int row = r + chalf;
                Pw[row * PL_STRIDE + (lane & 15)] =
                    __builtin_bit_cast(unsigned short, (__bf16)s0[r]);
                Pw[row * PL_STRIDE + 16 + (lane & 15)] =
                    __builtin_bit_cast(unsigned short, (__bf16)s1[r]);
            }
            // DS ops are in-order within a wave: safe to read back immediately.
            const int rowA = lane & 15;
            const int koff = (lane >> 4) * 8;
            v8bf lo = *(const v8bf*)&Pw[rowA * PL_STRIDE + koff];
            v8bf hi = *(const v8bf*)&Pw[rowA * PL_STRIDE + koff + 16];
            const v16bf pa = cat16(lo, hi);

            // row sums: acc_l += P @ ones  (replicated across the 16 columns)
            acc_l = __builtin_amdgcn_wmma_f32_16x16x32_bf16(
                false, pa, false, ones, (short)0, acc_l, false, false);

            // ---- P(16x32) @ V(32x64): 4 WMMA over dim tiles of 16 ---------
#pragma unroll
            for (int t = 0; t < 4; ++t) {
                const int dim = 16 * t + (lane & 15);
                const int kb = (lane >> 4) * 8;
                v8bf vlo = *(const v8bf*)&Vt[dim * VT_STRIDE + kb];
                v8bf vhi = *(const v8bf*)&Vt[dim * VT_STRIDE + kb + 16];
                v16bf vb = cat16(vlo, vhi);
                acc[t] = __builtin_amdgcn_wmma_f32_16x16x32_bf16(
                    false, pa, false, vb, (short)0, acc[t], false, false);
            }
        }
    }

    // ---- epilogue: normalize and store bf16 context -----------------------
    const int rhalf = (lane >> 4) * 8;
#pragma unroll
    for (int r = 0; r < 8; ++r) {
        const float inv = __builtin_amdgcn_rcpf(acc_l[r]);
        const size_t row = rowbase + q0 + r + rhalf;
#pragma unroll
        for (int t = 0; t < 4; ++t) {
            const int col = h * HDIM + 16 * t + (lane & 15);
            Ctx[row * EMB + col] = (bf16_t)(acc[t][r] * inv);
        }
    }
}

// ---------------------------------------------------------------------------
// Host launcher.
// Inputs: 0 query, 1 key, 2 value, 3 Wq, 4 bq, 5 Wk, 6 bk, 7 Wv, 8 bv,
//         9 Wo, 10 bo (all f32). Output: [B,S,E] f32.
// Workspace (48 MiB): Xbuf 8 | Wq..Wo 2*4 | Qp 8 | Kp 8 | Vp 8 | Ctx 8.
// ---------------------------------------------------------------------------
extern "C" void kernel_launch(void* const* d_in, const int* in_sizes, int n_in,
                              void* d_out, int out_size, void* d_ws,
                              size_t ws_size, hipStream_t stream) {
    const float* q  = (const float*)d_in[0];
    const float* k  = (const float*)d_in[1];
    const float* v  = (const float*)d_in[2];
    const float* Wf[4] = {(const float*)d_in[3], (const float*)d_in[5],
                          (const float*)d_in[7], (const float*)d_in[9]};
    const float* bq = (const float*)d_in[4];
    const float* bk = (const float*)d_in[6];
    const float* bv = (const float*)d_in[8];
    const float* bo = (const float*)d_in[10];
    float* out = (float*)d_out;

    const size_t XB = (size_t)MROWS * EMB * sizeof(bf16_t);   // 8 MiB
    const size_t WB = (size_t)EMB * EMB * sizeof(bf16_t);     // 2 MiB
    char* w = (char*)d_ws;
    bf16_t* Xbuf = (bf16_t*)(w);
    bf16_t* Wb[4];
    for (int i = 0; i < 4; ++i) Wb[i] = (bf16_t*)(w + XB + i * WB);
    bf16_t* Qp = (bf16_t*)(w + XB + 4 * WB);
    bf16_t* Kp = (bf16_t*)(w + 2 * XB + 4 * WB);
    bf16_t* Vp = (bf16_t*)(w + 3 * XB + 4 * WB);
    bf16_t* Cx = (bf16_t*)(w + 4 * XB + 4 * WB);

    const int nX = MROWS * EMB;       // 4.19M elems
    const int nW = EMB * EMB;         // 1.05M elems
    const dim3 cvtXGrid(nX / 8 / 256), cvtWGrid(nW / 8 / 256), cvtBlk(256);
    const dim3 gGrid(EMB / 128, MROWS / 128);   // (8, 32)
    const dim3 gBlk(256);

    // weights once
    for (int i = 0; i < 4; ++i)
        cvt_bf16_kernel<<<cvtWGrid, cvtBlk, 0, stream>>>(Wf[i], Wb[i], nW);

    // q/k/v share one staging buffer (stream-ordered, so safe)
    cvt_bf16_kernel<<<cvtXGrid, cvtBlk, 0, stream>>>(q, Xbuf, nX);
    gemm_xwt_kernel<true><<<gGrid, gBlk, 0, stream>>>(Xbuf, Wb[0], bq, Qp,
                                                      nullptr, MROWS, EMB, EMB);
    cvt_bf16_kernel<<<cvtXGrid, cvtBlk, 0, stream>>>(k, Xbuf, nX);
    gemm_xwt_kernel<true><<<gGrid, gBlk, 0, stream>>>(Xbuf, Wb[1], bk, Kp,
                                                      nullptr, MROWS, EMB, EMB);
    cvt_bf16_kernel<<<cvtXGrid, cvtBlk, 0, stream>>>(v, Xbuf, nX);
    gemm_xwt_kernel<true><<<gGrid, gBlk, 0, stream>>>(Xbuf, Wb[2], bv, Vp,
                                                      nullptr, MROWS, EMB, EMB);

    const dim3 aGrid(SEQ / 64, NHEAD, BATCH);   // (32, 16, 2)
    attention_kernel<<<aGrid, dim3(128), 0, stream>>>(Qp, Kp, Vp, Cx);

    gemm_xwt_kernel<false><<<gGrid, gBlk, 0, stream>>>(Cx, Wb[3], bo, nullptr,
                                                       out, MROWS, EMB, EMB);
}
